// MultiomicTransformer_76828374991777
// MI455X (gfx1250) — matrix operand order repaired and checked by reading, not compiled, not run on hardware
//
#include <hip/hip_runtime.h>
#include <math.h>

// ---------------------------------------------------------------------------
// Types for CDNA5 WMMA (wave32, 16x16x32 bf16 -> f32)
// ---------------------------------------------------------------------------
typedef __attribute__((ext_vector_type(16))) __bf16 v16bf;
typedef __attribute__((ext_vector_type(8)))  __bf16 v8bf;
typedef __attribute__((ext_vector_type(4)))  __bf16 v4bf;
typedef __attribute__((ext_vector_type(8)))  float  v8f;

union U16BF { v16bf v; v8bf h[2]; };

__device__ __forceinline__ v8f wmma_bf16(v16bf a, v16bf b, v8f c) {
    // D = A(16x32 bf16) * B(32x16 bf16) + C(16x16 f32)
    return __builtin_amdgcn_wmma_f32_16x16x32_bf16(
        false, a, false, b, (short)0, c, false, false);
}

__device__ __forceinline__ v4bf cvt4(float4 f) {
    v4bf r;
    r[0] = (__bf16)f.x; r[1] = (__bf16)f.y;
    r[2] = (__bf16)f.z; r[3] = (__bf16)f.w;
    return r;
}

// A-matrix fragment (16x32 bf16): lane&15 = M row; lane>=16 selects K half.
// elems 0..7 -> K = kk + kh + 0..7 ; elems 8..15 -> K = kk + 16 + kh + 0..7
__device__ __forceinline__ v16bf ld_afrag(const __bf16* rowbase, int pitch,
                                          int kk, int lane) {
    int m  = lane & 15;
    int kh = (lane >> 4) * 8;
    const __bf16* p = rowbase + (size_t)m * pitch + kk + kh;
    U16BF u;
    u.h[0] = *(const v8bf*)(p);
    u.h[1] = *(const v8bf*)(p + 16);
    return u.v;
}

// B-matrix fragment (32x16 bf16): lane&15 = N col; lane>=16 selects K half.
// elems 0..7 -> K = kh + 0..7 ; elems 8..15 -> K = 16 + kh + 0..7
__device__ __forceinline__ v16bf ld_bfrag(const __bf16* kbase, int pitch,
                                          int n0, int lane) {
    int n  = lane & 15;
    int kh = (lane >> 4) * 8;
    const __bf16* p = kbase + n0 + n;
    v16bf v;
#pragma unroll
    for (int s = 0; s < 8; ++s) {
        v[s]     = p[(size_t)(kh + s) * pitch];
        v[s + 8] = p[(size_t)(16 + kh + s) * pitch];
    }
    return v;
}

// ---------------------------------------------------------------------------
// Generic GEMM: C[M,N] = res + s * act( A[M,K] @ W[N,K]^T + bias[N] )
// 64x64 tile per 256-thread block; 8 waves, 2 WMMA tiles each; BK = 64.
// N % 64 == 0, K % 64 == 0; M arbitrary (masked).
// act: 0 none, 1 relu, 2 tanh
// ---------------------------------------------------------------------------
__global__ __launch_bounds__(256) void k_gemm(
    const float* __restrict__ A, const float* __restrict__ Wt,
    const float* __restrict__ bias, const float* __restrict__ res,
    const float* __restrict__ scale_ptr, float scale_c,
    float* __restrict__ C, int M, int N, int Kd, int act)
{
    const int tid  = threadIdx.x;
    const int lane = tid & 31;
    const int wv   = tid >> 5;
    const int mt   = wv >> 1;          // 0..3  (16-row slab)
    const int nt   = (wv & 1) * 32;    // 0 / 32

    const int n0 = blockIdx.x * 64;
    const int m0 = blockIdx.y * 64;

    __shared__ __align__(16) __bf16 As[64][72];   // 64x64 bf16, pad->72
    __shared__ __align__(16) __bf16 Bs[64][72];   // B[k][n] (transposed W)

    v8f acc0 = {}, acc1 = {};

    for (int k0 = 0; k0 < Kd; k0 += 64) {
        __syncthreads();
        // ---- stage: vectorized float4 loads, batched before conversion ----
        float4 va[4], vb[4];
#pragma unroll
        for (int i = 0; i < 4; ++i) {
            int vec = tid + i * 256;           // 0..1023
            int r   = vec >> 4;                // row (m or n), 0..63
            int kv  = (vec & 15) * 4;          // k offset, 0..60
            int gm  = m0 + r;
            int gmc = gm < M ? gm : (M - 1);
            va[i] = *(const float4*)(A + (size_t)gmc * Kd + k0 + kv);
            if (gm >= M) va[i] = make_float4(0.f, 0.f, 0.f, 0.f);
            vb[i] = *(const float4*)(Wt + (size_t)(n0 + r) * Kd + k0 + kv);
        }
        if (k0 + 64 < Kd) {
            // pull next k-tile toward the caches (global_prefetch_b8)
            int r = tid >> 4, kv = (tid & 15) * 4;
            int gm = m0 + r;
            if (gm < M) __builtin_prefetch(A + (size_t)gm * Kd + k0 + 64 + kv, 0, 0);
            __builtin_prefetch(Wt + (size_t)(n0 + r) * Kd + k0 + 64 + kv, 0, 0);
        }
#pragma unroll
        for (int i = 0; i < 4; ++i) {
            int vec = tid + i * 256;
            int r   = vec >> 4;
            int kv  = (vec & 15) * 4;
            *(v4bf*)&As[r][kv] = cvt4(va[i]);                 // ds_store_b64
            Bs[kv + 0][r] = (__bf16)vb[i].x;                  // transpose
            Bs[kv + 1][r] = (__bf16)vb[i].y;
            Bs[kv + 2][r] = (__bf16)vb[i].z;
            Bs[kv + 3][r] = (__bf16)vb[i].w;
        }
        __syncthreads();

#pragma unroll
        for (int kk = 0; kk < 64; kk += 32) {
            v16bf af = ld_afrag(&As[mt * 16][0], 72, kk, lane);
            v16bf b0 = ld_bfrag(&Bs[kk][0], 72, nt, lane);
            v16bf b1 = ld_bfrag(&Bs[kk][0], 72, nt + 16, lane);
            acc0 = wmma_bf16(af, b0, acc0);
            acc1 = wmma_bf16(af, b1, acc1);
        }
    }

    const float s = scale_c * (scale_ptr ? scale_ptr[0] : 1.0f);
    const int rbase = mt * 16 + ((lane >= 16) ? 8 : 0);
    const int col   = lane & 15;
#pragma unroll
    for (int r = 0; r < 8; ++r) {
        int gm = m0 + rbase + r;
        if (gm >= M) continue;
        float v0 = acc0[r], v1 = acc1[r];
        int gn0 = n0 + nt + col, gn1 = n0 + nt + 16 + col;
        if (bias) { v0 += bias[gn0]; v1 += bias[gn1]; }
        if (act == 1) { v0 = v0 > 0.f ? v0 : 0.f; v1 = v1 > 0.f ? v1 : 0.f; }
        else if (act == 2) { v0 = tanhf(v0); v1 = tanhf(v1); }
        v0 *= s; v1 *= s;
        if (res) { v0 += res[(size_t)gm * N + gn0]; v1 += res[(size_t)gm * N + gn1]; }
        C[(size_t)gm * N + gn0] = v0;
        C[(size_t)gm * N + gn1] = v1;
    }
}

// ---------------------------------------------------------------------------
// Flash attention, dk = 64, heads interleaved in last dim (row stride ld*).
// One block = 64 query rows of one (b,h). Streams K/V in 64-tiles with
// online softmax; QK^T and P@V both via WMMA bf16. Optional bias[b,q,s].
// Lq % 64 == 0, Lkv % 64 == 0.
// ---------------------------------------------------------------------------
#define DK 64
__global__ __launch_bounds__(256) void k_attn(
    const float* __restrict__ Q, int ldq,
    const float* __restrict__ K, int ldk,
    const float* __restrict__ V, int ldv,
    float* __restrict__ O, int ldo,
    const float* __restrict__ bias, float bias_scale,
    int Lq, int Lkv, int H, float scale)
{
    const int bh = blockIdx.x;
    const int b  = bh / H;
    const int h  = bh % H;
    const int q0 = blockIdx.y * 64;

    const int tid  = threadIdx.x;
    const int lane = tid & 31;
    const int wv   = tid >> 5;
    const int mt   = wv >> 1;
    const int nt   = (wv & 1) * 32;

    __shared__ __align__(16) __bf16 Qs[64][72];
    __shared__ __align__(16) __bf16 Ks[64][72];   // transposed: Ks[d][s]
    __shared__ __align__(16) __bf16 Vs[64][72];   // Vs[s][d]
    __shared__ __align__(16) __bf16 Pb[64][72];
    __shared__ float Ps[64][65];
    __shared__ float red4[64][4];
    __shared__ float rowm[64], rowl[64], rowa[64];

    // ---- stage Q tile (vectorized) ----
    const float* qb = Q + (size_t)b * Lq * ldq + (size_t)h * DK;
    {
        float4 vq[4];
#pragma unroll
        for (int i = 0; i < 4; ++i) {
            int vec = tid + i * 256;
            int m = vec >> 4, dv = (vec & 15) * 4;
            vq[i] = *(const float4*)(qb + (size_t)(q0 + m) * ldq + dv);
        }
#pragma unroll
        for (int i = 0; i < 4; ++i) {
            int vec = tid + i * 256;
            int m = vec >> 4, dv = (vec & 15) * 4;
            *(v4bf*)&Qs[m][dv] = cvt4(vq[i]);
        }
    }
    if (tid < 64) { rowm[tid] = -1e30f; rowl[tid] = 0.0f; }

    v8f o0 = {}, o1 = {};

    const float* kb = K + (size_t)b * Lkv * ldk + (size_t)h * DK;
    const float* vb = V + (size_t)b * Lkv * ldv + (size_t)h * DK;

    for (int st = 0; st < Lkv; st += 64) {
        __syncthreads();
        // ---- stage K (transposed) and V, vectorized ----
        {
            float4 vk[4], vvv[4];
#pragma unroll
            for (int i = 0; i < 4; ++i) {
                int vec = tid + i * 256;
                int s = vec >> 4, dv = (vec & 15) * 4;
                vk[i]  = *(const float4*)(kb + (size_t)(st + s) * ldk + dv);
                vvv[i] = *(const float4*)(vb + (size_t)(st + s) * ldv + dv);
            }
            if (st + 64 < Lkv) {
                int s = tid >> 2, dv = (tid & 3) * 16;
                __builtin_prefetch(kb + (size_t)(st + 64 + s) * ldk + dv, 0, 0);
                __builtin_prefetch(vb + (size_t)(st + 64 + s) * ldv + dv, 0, 0);
            }
#pragma unroll
            for (int i = 0; i < 4; ++i) {
                int vec = tid + i * 256;
                int s = vec >> 4, dv = (vec & 15) * 4;
                Ks[dv + 0][s] = (__bf16)vk[i].x;
                Ks[dv + 1][s] = (__bf16)vk[i].y;
                Ks[dv + 2][s] = (__bf16)vk[i].z;
                Ks[dv + 3][s] = (__bf16)vk[i].w;
                *(v4bf*)&Vs[s][dv] = cvt4(vvv[i]);
            }
        }
        __syncthreads();

        // ---- S = Q @ K^T  (per wave: 16x32 sub-tile) ----
        v8f a0 = {}, a1 = {};
#pragma unroll
        for (int kk = 0; kk < 64; kk += 32) {
            v16bf af = ld_afrag(&Qs[mt * 16][0], 72, kk, lane);
            v16bf b0 = ld_bfrag(&Ks[kk][0], 72, nt, lane);
            v16bf b1 = ld_bfrag(&Ks[kk][0], 72, nt + 16, lane);
            a0 = wmma_bf16(af, b0, a0);
            a1 = wmma_bf16(af, b1, a1);
        }
        {
            const int rbase = mt * 16 + ((lane >= 16) ? 8 : 0);
            const int col   = lane & 15;
#pragma unroll
            for (int r = 0; r < 8; ++r) {
                int m = rbase + r;
                float s0v = a0[r] * scale;
                float s1v = a1[r] * scale;
                if (bias) {
                    const float* bp = bias + ((size_t)b * Lq + (q0 + m)) * (size_t)Lkv + st;
                    s0v += bias_scale * bp[nt + col];
                    s1v += bias_scale * bp[nt + 16 + col];
                }
                Ps[m][nt + col]      = s0v;
                Ps[m][nt + 16 + col] = s1v;
            }
        }
        __syncthreads();

        // ---- online softmax, 4 threads per row ----
        {
            int r = tid >> 2, sg = tid & 3;
            float mx = -1e30f;
#pragma unroll
            for (int j = 0; j < 16; ++j) mx = fmaxf(mx, Ps[r][sg * 16 + j]);
            red4[r][sg] = mx;
            __syncthreads();
            float mnew = fmaxf(fmaxf(red4[r][0], red4[r][1]),
                               fmaxf(red4[r][2], red4[r][3]));
            mnew = fmaxf(mnew, rowm[r]);
            float lsum = 0.0f;
#pragma unroll
            for (int j = 0; j < 16; ++j) {
                float p = __expf(Ps[r][sg * 16 + j] - mnew);
                Pb[r][sg * 16 + j] = (__bf16)p;
                lsum += p;
            }
            __syncthreads();               // red4 consumed by all before reuse
            red4[r][sg] = lsum;
            __syncthreads();
            if (sg == 0) {
                float ssum = red4[r][0] + red4[r][1] + red4[r][2] + red4[r][3];
                float al = __expf(rowm[r] - mnew);
                rowl[r] = rowl[r] * al + ssum;
                rowm[r] = mnew;
                rowa[r] = al;
            }
        }
        __syncthreads();

        // ---- rescale O and accumulate P @ V ----
        {
            const int rbase = mt * 16 + ((lane >= 16) ? 8 : 0);
#pragma unroll
            for (int r = 0; r < 8; ++r) {
                float al = rowa[rbase + r];
                o0[r] *= al; o1[r] *= al;
            }
        }
#pragma unroll
        for (int kk = 0; kk < 64; kk += 32) {
            v16bf af = ld_afrag(&Pb[mt * 16][0], 72, kk, lane);
            v16bf b0 = ld_bfrag(&Vs[kk][0], 72, nt, lane);
            v16bf b1 = ld_bfrag(&Vs[kk][0], 72, nt + 16, lane);
            o0 = wmma_bf16(af, b0, o0);
            o1 = wmma_bf16(af, b1, o1);
        }
    }
    __syncthreads();

    float* ob = O + (size_t)b * Lq * ldo + (size_t)h * DK;
    const int rbase = mt * 16 + ((lane >= 16) ? 8 : 0);
    const int col   = lane & 15;
#pragma unroll
    for (int r = 0; r < 8; ++r) {
        int m = rbase + r;
        float inv = 1.0f / rowl[m];
        ob[(size_t)(q0 + m) * ldo + nt + col]      = o0[r] * inv;
        ob[(size_t)(q0 + m) * ldo + nt + 16 + col] = o1[r] * inv;
    }
}

// ---------------------------------------------------------------------------
// LayerNorm over D=512 rows. One wave per row (16 elems/lane, shfl reduce).
// Y = LN(X (+R2)) * g + b (+ post[row % post_rows])
// ---------------------------------------------------------------------------
__global__ __launch_bounds__(256) void k_ln(
    const float* __restrict__ X, const float* __restrict__ R2,
    const float* __restrict__ g, const float* __restrict__ bta,
    const float* __restrict__ post, int post_rows,
    float* __restrict__ Y, int rows)
{
    int row = blockIdx.x * 8 + (threadIdx.x >> 5);
    if (row >= rows) return;
    int lane = threadIdx.x & 31;
    const float* x  = X + (size_t)row * 512;
    const float* r2 = R2 ? R2 + (size_t)row * 512 : nullptr;

    float v[16];
    float s = 0.0f;
#pragma unroll
    for (int i = 0; i < 16; ++i) {
        float t = x[lane + 32 * i];
        if (r2) t += r2[lane + 32 * i];
        v[i] = t; s += t;
    }
#pragma unroll
    for (int m = 16; m > 0; m >>= 1) s += __shfl_xor(s, m, 32);
    float mu = s * (1.0f / 512.0f);
    float vs = 0.0f;
#pragma unroll
    for (int i = 0; i < 16; ++i) { float d = v[i] - mu; vs += d * d; }
#pragma unroll
    for (int m = 16; m > 0; m >>= 1) vs += __shfl_xor(vs, m, 32);
    float rs = rsqrtf(vs * (1.0f / 512.0f) + 1e-5f);

    const float* pp = post ? post + (size_t)(row % post_rows) * 512 : nullptr;
#pragma unroll
    for (int i = 0; i < 16; ++i) {
        int c = lane + 32 * i;
        float y = (v[i] - mu) * rs * g[c] + bta[c];
        if (pp) y += pp[c];
        Y[(size_t)row * 512 + c] = y;
    }
}

// ---------------------------------------------------------------------------
// H[row][f] = relu(x[row] * w1[f]), F = 2048
// ---------------------------------------------------------------------------
__global__ void k_buildh(const float* __restrict__ x, const float* __restrict__ w1,
                         float* __restrict__ H, long total)
{
    long i = (long)blockIdx.x * 256 + threadIdx.x;
    if (i >= total) return;
    long row = i >> 11;
    int f = (int)(i & 2047);
    float v = x[row] * w1[f];
    H[i] = v > 0.0f ? v : 0.0f;
}

// sinusoidal positional encoding [1024, 512]
__global__ void k_pos(float* __restrict__ P)
{
    int i = blockIdx.x * 256 + threadIdx.x;
    int w = i >> 9, d = i & 511;
    const float lg = 9.210340371976184f; // ln(10000)
    if (d < 256) {
        float f = __expf(-lg * (float)d * (1.0f / 256.0f));
        P[i] = sinf((float)w * f);
    } else {
        float f = __expf(-lg * (float)(d - 256) * (1.0f / 256.0f));
        P[i] = cosf((float)w * f);
    }
}

// gather rows: out[row] = emb[ids[row % idmod]]
__global__ void k_gather(const float* __restrict__ emb, const int* __restrict__ ids,
                         float* __restrict__ out, long rows, int idmod)
{
    long i = (long)blockIdx.x * 256 + threadIdx.x;
    if (i >= rows * 512) return;
    long row = i >> 9;
    int d = (int)(i & 511);
    out[i] = emb[(size_t)ids[row % idmod] * 512 + d];
}

// concat [4,512]+[4,512] -> [4,1024]
__global__ void k_concat(const float* __restrict__ a, const float* __restrict__ b,
                         float* __restrict__ c)
{
    int i = blockIdx.x * 256 + threadIdx.x;
    int r = i >> 10, d = i & 1023;
    c[i] = (d < 512) ? a[r * 512 + d] : b[r * 512 + d - 512];
}

// attention pooling: out[b] = sum_l softmax_l(<x_l,q>/sqrt(D)) x_l
__global__ __launch_bounds__(256) void k_pool(
    const float* __restrict__ X, const float* __restrict__ q,
    float* __restrict__ out, int L)
{
    int b = blockIdx.x, tid = threadIdx.x;
    __shared__ float sc[1024];
    __shared__ float red[256];
    const float* xb = X + (size_t)b * L * 512;
    for (int l = tid; l < L; l += 256) {
        float s = 0.0f;
        for (int d = 0; d < 512; ++d) s += xb[(size_t)l * 512 + d] * q[d];
        sc[l] = s * 0.044194173824159216f; // 1/sqrt(512)
    }
    __syncthreads();
    float mx = -1e30f;
    for (int l = tid; l < L; l += 256) mx = fmaxf(mx, sc[l]);
    red[tid] = mx;
    __syncthreads();
    for (int s = 128; s > 0; s >>= 1) {
        if (tid < s) red[tid] = fmaxf(red[tid], red[tid + s]);
        __syncthreads();
    }
    mx = red[0];
    __syncthreads();
    float sum = 0.0f;
    for (int l = tid; l < L; l += 256) {
        float e = __expf(sc[l] - mx);
        sc[l] = e; sum += e;
    }
    red[tid] = sum;
    __syncthreads();
    for (int s = 128; s > 0; s >>= 1) {
        if (tid < s) red[tid] += red[tid + s];
        __syncthreads();
    }
    float inv = 1.0f / red[0];
    __syncthreads();
    for (int d = tid; d < 512; d += 256) {
        float a = 0.0f;
        for (int l = 0; l < L; ++l) a += sc[l] * xb[(size_t)l * 512 + d];
        out[(size_t)b * 512 + d] = a * inv;
    }
}

// pred[row] = dot(TG[row] + CTX[row/G], gw) + gb
__global__ __launch_bounds__(256) void k_pred(
    const float* __restrict__ TG, const float* __restrict__ CTX,
    const float* __restrict__ gw, const float* __restrict__ gb,
    float* __restrict__ out, int G)
{
    int row = blockIdx.x * 8 + (threadIdx.x >> 5);
    int lane = threadIdx.x & 31;
    int b = row / G;
    const float* t = TG + (size_t)row * 512;
    const float* c = CTX + (size_t)b * 512;
    float s = 0.0f;
#pragma unroll
    for (int i = 0; i < 16; ++i) {
        int k = lane + 32 * i;
        s += (t[k] + c[k]) * gw[k];
    }
#pragma unroll
    for (int m = 16; m > 0; m >>= 1) s += __shfl_xor(s, m, 32);
    if (lane == 0) out[row] = s + gb[0];
}

// per-row (T=512): softmax -> top-16 mask -> renormalize; in-place
__global__ __launch_bounds__(256) void k_topk(float* __restrict__ A)
{
    int g = blockIdx.x, tid = threadIdx.x;
    __shared__ float a[512];
    __shared__ int   sel[512];
    __shared__ float rv[256];
    __shared__ int   ri[256];
    float* row = A + (size_t)g * 512;

    a[tid] = row[tid]; a[tid + 256] = row[tid + 256];
    sel[tid] = 0; sel[tid + 256] = 0;
    __syncthreads();

    // softmax
    rv[tid] = fmaxf(a[tid], a[tid + 256]);
    __syncthreads();
    for (int s = 128; s > 0; s >>= 1) {
        if (tid < s) rv[tid] = fmaxf(rv[tid], rv[tid + s]);
        __syncthreads();
    }
    float mx = rv[0];
    __syncthreads();
    float e1 = __expf(a[tid] - mx), e2 = __expf(a[tid + 256] - mx);
    a[tid] = e1; a[tid + 256] = e2;
    rv[tid] = e1 + e2;
    __syncthreads();
    for (int s = 128; s > 0; s >>= 1) {
        if (tid < s) rv[tid] += rv[tid + s];
        __syncthreads();
    }
    float inv = 1.0f / rv[0];
    __syncthreads();
    a[tid] *= inv; a[tid + 256] *= inv;
    __syncthreads();

    // select top-16 by repeated argmax
    for (int it = 0; it < 16; ++it) {
        float bv = -1.0f; int bi = tid;
        if (!sel[tid]) { bv = a[tid]; bi = tid; }
        if (!sel[tid + 256] && a[tid + 256] > bv) { bv = a[tid + 256]; bi = tid + 256; }
        rv[tid] = bv; ri[tid] = bi;
        __syncthreads();
        for (int s = 128; s > 0; s >>= 1) {
            if (tid < s && rv[tid + s] > rv[tid]) { rv[tid] = rv[tid + s]; ri[tid] = ri[tid + s]; }
            __syncthreads();
        }
        if (tid == 0) sel[ri[0]] = 1;
        __syncthreads();
    }

    float ms = (sel[tid] ? a[tid] : 0.0f) + (sel[tid + 256] ? a[tid + 256] : 0.0f);
    rv[tid] = ms;
    __syncthreads();
    for (int s = 128; s > 0; s >>= 1) {
        if (tid < s) rv[tid] += rv[tid + s];
        __syncthreads();
    }
    float inv2 = 1.0f / (rv[0] + 1e-8f);
    __syncthreads();
    row[tid]       = sel[tid]       ? a[tid] * inv2       : 0.0f;
    row[tid + 256] = sel[tid + 256] ? a[tid + 256] * inv2 : 0.0f;
}

// ---------------------------------------------------------------------------
// Orchestration
// ---------------------------------------------------------------------------
extern "C" void kernel_launch(void* const* d_in, const int* in_sizes, int n_in,
                              void* d_out, int out_size, void* d_ws, size_t ws_size,
                              hipStream_t stream)
{
    (void)in_sizes; (void)n_in; (void)out_size; (void)ws_size;
    const int Bc = 4, Wn = 1024, Tn = 512, Gn = 2048, NH = 8;
    const float ATTN_SC = 0.125f;               // 1/sqrt(64)
    const float INV_SQRT_D = 0.044194173824159216f; // 1/sqrt(512)

    auto F = [&](int i) { return (const float*)d_in[i]; };
    const float* atac    = F(0);
    const float* tf_expr = F(1);
    const float* bias    = F(2);
    const int*   tf_ids  = (const int*)d_in[3];
    const int*   tg_ids  = (const int*)d_in[4];
    const float* tf_emb  = F(5);
    const float* tg_emb  = F(6);
    const float* tg_dec  = F(7);
    const float* atac_w1 = F(8);
    const float* atac_w2 = F(9);
    const float* atac_ln_g = F(10); const float* atac_ln_b = F(11);
    const float* tf_w1 = F(12); const float* tf_w2 = F(13);
    const float* tf_ln_g = F(14); const float* tf_ln_b = F(15);
    const int ENC = 16;  // 12 tensors per layer x 4
    const int C1 = 64, C2 = 74, C3 = 84; // cross-attn param bases
    const float* tf_pool_q   = F(94);
    const float* atac_pool_q = F(95);
    const float* out_w1 = F(96); const float* out_w2 = F(97);
    const float* out_ln_g = F(98); const float* out_ln_b = F(99);
    const float* gene_w = F(100); const float* gene_b = F(101);
    const float* sc_scale = F(102);

    float* out = (float*)d_out;

    // workspace arena (floats)
    float* ws = (float*)d_ws;
    size_t off = 0;
    auto alloc = [&](size_t n) { float* p = ws + off; off += n; return p; };
    float* XW  = alloc((size_t)4096 * 512);   // encoder state x
    float* HB  = alloc((size_t)4096 * 512);   // LN scratch
    float* CC  = alloc((size_t)4096 * 2048);  // big scratch (H / qkv / ff1 / sim)
    float* DD  = alloc((size_t)2048 * 512);   // tf_emb (pre-cross)
    float* TFC = alloc((size_t)2048 * 512);   // tf_cross
    float* EE  = alloc((size_t)8192 * 512);   // Q proj
    float* FFb = alloc((size_t)4096 * 512);   // K proj
    float* GG  = alloc((size_t)4096 * 512);   // V proj
    float* HH  = alloc((size_t)8192 * 512);   // attention out
    float* JJ  = alloc((size_t)8192 * 512);   // tg state
    float* II  = alloc((size_t)4096 * 512);   // atac_cross
    float* POS = alloc((size_t)1024 * 512);
    float* TFB = alloc((size_t)512 * 512);    // tf_emb[tf_ids]
    float* TGD = alloc((size_t)2048 * 512);   // tg_dec[tg_ids]
    float* TFP = alloc(4 * 512);
    float* AP  = alloc(4 * 512);
    float* CAT = alloc(4 * 1024);
    float* CX1 = alloc(4 * 2048);
    float* CTX = alloc(4 * 512);

    auto gemm = [&](const float* A, const float* Wt, const float* bi,
                    const float* res, const float* sp, float sc,
                    float* C, int M, int N, int K, int act) {
        dim3 g(N / 64, (M + 63) / 64);
        k_gemm<<<g, 256, 0, stream>>>(A, Wt, bi, res, sp, sc, C, M, N, K, act);
    };
    auto attn = [&](const float* Q, int ldq, const float* K, int ldk,
                    const float* V, int ldv, float* O, int ldo,
                    const float* bp, float bsc, int Lq, int Lkv) {
        dim3 g(Bc * NH, Lq / 64);
        k_attn<<<g, 256, 0, stream>>>(Q, ldq, K, ldk, V, ldv, O, ldo,
                                      bp, bsc, Lq, Lkv, NH, ATTN_SC);
    };
    auto ln = [&](const float* X, const float* R2, const float* g,
                  const float* b, const float* post, int prows,
                  float* Y, int rows) {
        k_ln<<<(rows + 7) / 8, 256, 0, stream>>>(X, R2, g, b, post, prows, Y, rows);
    };

    // ---- ATAC stream ----
    k_pos<<<(1024 * 512) / 256, 256, 0, stream>>>(POS);
    k_buildh<<<(4096L * 2048) / 256, 256, 0, stream>>>(atac, atac_w1, CC, 4096L * 2048);
    gemm(CC, atac_w2, nullptr, nullptr, nullptr, 1.f, HB, 4096, 512, 2048, 0);
    ln(HB, nullptr, atac_ln_g, atac_ln_b, POS, 1024, XW, 4096);

    // ---- encoder x4 (norm_first) ----
    for (int L = 0; L < 4; ++L) {
        int b0 = ENC + 12 * L;
        const float* in_w = F(b0), *in_b = F(b0 + 1);
        const float* out_w = F(b0 + 2), *out_b = F(b0 + 3);
        const float* l1w = F(b0 + 4), *l1b = F(b0 + 5);
        const float* l2w = F(b0 + 6), *l2b = F(b0 + 7);
        const float* g1 = F(b0 + 8), *be1 = F(b0 + 9);
        const float* g2 = F(b0 + 10), *be2 = F(b0 + 11);

        ln(XW, nullptr, g1, be1, nullptr, 0, HB, 4096);
        gemm(HB, in_w, in_b, nullptr, nullptr, 1.f, CC, 4096, 1536, 512, 0);
        attn(CC, 1536, CC + 512, 1536, CC + 1024, 1536, HH, 512,
             nullptr, 0.f, Wn, Wn);
        gemm(HH, out_w, out_b, XW, nullptr, 1.f, XW, 4096, 512, 512, 0);
        ln(XW, nullptr, g2, be2, nullptr, 0, HB, 4096);
        gemm(HB, l1w, l1b, nullptr, nullptr, 1.f, CC, 4096, 2048, 512, 1);
        gemm(CC, l2w, l2b, XW, nullptr, 1.f, XW, 4096, 512, 2048, 0);
    }

    // ---- TF stream ----
    k_buildh<<<(2048L * 2048) / 256, 256, 0, stream>>>(tf_expr, tf_w1, CC, 2048L * 2048);
    gemm(CC, tf_w2, nullptr, nullptr, nullptr, 1.f, HH, 2048, 512, 2048, 0);
    k_gather<<<(512L * 512) / 256, 256, 0, stream>>>(tf_emb, tf_ids, TFB, 512, 512);
    ln(HH, nullptr, tf_ln_g, tf_ln_b, TFB, 512, DD, 2048);

    // ---- tf2atac cross: LN(tf_emb + MHA(tf_emb, win)) ----
    {
        const float* wq = F(C1), *bq = F(C1 + 1), *wk = F(C1 + 2), *bk = F(C1 + 3);
        const float* wv = F(C1 + 4), *bv = F(C1 + 5), *wo = F(C1 + 6), *bo = F(C1 + 7);
        const float* lg = F(C1 + 8), *lb = F(C1 + 9);
        gemm(DD, wq, bq, nullptr, nullptr, 1.f, EE, 2048, 512, 512, 0);
        gemm(XW, wk, bk, nullptr, nullptr, 1.f, FFb, 4096, 512, 512, 0);
        gemm(XW, wv, bv, nullptr, nullptr, 1.f, GG, 4096, 512, 512, 0);
        attn(EE, 512, FFb, 512, GG, 512, HH, 512, nullptr, 0.f, Tn, Wn);
        gemm(HH, wo, bo, DD, nullptr, 1.f, TFC, 2048, 512, 512, 0);
        ln(TFC, nullptr, lg, lb, nullptr, 0, TFC, 2048);
    }

    // ---- atac2tf cross: LN(win + MHA(win, tf_emb)) ----
    {
        const float* wq = F(C2), *bq = F(C2 + 1), *wk = F(C2 + 2), *bk = F(C2 + 3);
        const float* wv = F(C2 + 4), *bv = F(C2 + 5), *wo = F(C2 + 6), *bo = F(C2 + 7);
        const float* lg = F(C2 + 8), *lb = F(C2 + 9);
        gemm(XW, wq, bq, nullptr, nullptr, 1.f, EE, 4096, 512, 512, 0);
        gemm(DD, wk, bk, nullptr, nullptr, 1.f, FFb, 2048, 512, 512, 0);
        gemm(DD, wv, bv, nullptr, nullptr, 1.f, GG, 2048, 512, 512, 0);
        attn(EE, 512, FFb, 512, GG, 512, HH, 512, nullptr, 0.f, Wn, Tn);
        gemm(HH, wo, bo, XW, nullptr, 1.f, II, 4096, 512, 512, 0);
        ln(II, nullptr, lg, lb, nullptr, 0, II, 4096);
    }

    // ---- tg2atac cross with distance bias ----
    {
        const float* wq = F(C3), *bq = F(C3 + 1), *wk = F(C3 + 2), *bk = F(C3 + 3);
        const float* wv = F(C3 + 4), *bv = F(C3 + 5), *wo = F(C3 + 6), *bo = F(C3 + 7);
        const float* lg = F(C3 + 8), *lb = F(C3 + 9);
        k_gather<<<(8192L * 512) / 256, 256, 0, stream>>>(tg_emb, tg_ids, JJ, 8192, Gn);
        gemm(JJ, wq, bq, nullptr, nullptr, 1.f, EE, 8192, 512, 512, 0);
        gemm(II, wk, bk, nullptr, nullptr, 1.f, FFb, 4096, 512, 512, 0);
        gemm(II, wv, bv, nullptr, nullptr, 1.f, GG, 4096, 512, 512, 0);
        attn(EE, 512, FFb, 512, GG, 512, HH, 512, bias, 2.0f, Gn, Wn);
        gemm(HH, wo, bo, JJ, nullptr, 1.f, JJ, 8192, 512, 512, 0);
        ln(JJ, nullptr, lg, lb, nullptr, 0, JJ, 8192);
    }

    // ---- pooling + ctx ----
    k_pool<<<4, 256, 0, stream>>>(TFC, tf_pool_q, TFP, Tn);
    k_pool<<<4, 256, 0, stream>>>(II, atac_pool_q, AP, Wn);
    k_concat<<<(4 * 1024) / 256, 256, 0, stream>>>(TFP, AP, CAT);
    gemm(CAT, out_w1, nullptr, nullptr, nullptr, 1.f, CX1, 4, 2048, 1024, 2);
    gemm(CX1, out_w2, nullptr, nullptr, nullptr, 1.f, CTX, 4, 512, 2048, 0);
    ln(CTX, nullptr, out_ln_g, out_ln_b, nullptr, 0, CTX, 4);

    // ---- prediction head ----
    k_pred<<<8192 / 8, 256, 0, stream>>>(JJ, CTX, gene_w, gene_b, out, Gn);

    // ---- TF -> TG top-k shortcut ----
    k_gather<<<(2048L * 512) / 256, 256, 0, stream>>>(tg_dec, tg_ids, TGD, 2048, Gn);
    gemm(TGD, TFB, nullptr, nullptr, nullptr, INV_SQRT_D, CC, 2048, 512, 512, 0);
    k_topk<<<2048, 256, 0, stream>>>(CC);
    // pred += sc_scale * tf_expr @ attn^T   (attn is [G,T] => transB GEMM)
    gemm(tf_expr, CC, nullptr, out, sc_scale, 1.f, out, 4, 2048, 512, 0);
}
